// MultiHeadAttention_32418413150522
// MI455X (gfx1250) — compile-verified, matrix operand order checked by reading, not run on hardware
//
#include <hip/hip_runtime.h>
#include <math.h>

// ---------------------------------------------------------------------------
// MI455X (gfx1250) attention pipeline: bf16 WMMA, f32 accumulation,
// double-buffered LDS pipeline, async global->LDS staging where available.
// ---------------------------------------------------------------------------

typedef __bf16 bf16_t;
typedef __attribute__((ext_vector_type(16))) __bf16 v16bf;
typedef __attribute__((ext_vector_type(8)))  __bf16 v8bf;
typedef __attribute__((ext_vector_type(4)))  __bf16 v4bf;
typedef __attribute__((ext_vector_type(8)))  float  v8f;
typedef __attribute__((ext_vector_type(4)))  float  v4f;

#define B_ 16
#define N_ 2048
#define D_ 512
#define E_ 512
#define M_ (B_ * N_)   // 32768 tokens

#if defined(__has_builtin) && __has_builtin(__builtin_amdgcn_global_load_async_to_lds_b128)
#define HAVE_ASYNC_LDS 1
#else
#define HAVE_ASYNC_LDS 0
#endif

#if HAVE_ASYNC_LDS
// Builtin signature (from toolchain diagnostic): param0 = AS(1) int4*,
// param1 = AS(3) int4*, then two imm ints (offset, cpol).
typedef int v4i_ __attribute__((ext_vector_type(4)));
typedef __attribute__((address_space(1))) v4i_ gm_v4i;
typedef __attribute__((address_space(3))) v4i_ lds_v4i;

__device__ __forceinline__ void wait_async0() {
#if defined(__has_builtin) && __has_builtin(__builtin_amdgcn_s_wait_asynccnt)
    __builtin_amdgcn_s_wait_asynccnt(0);
#else
    asm volatile("s_wait_asynccnt 0x0" ::: "memory");
#endif
}
#endif

__device__ __forceinline__ v8f wmma_bf16(v16bf a, v16bf b, v8f c) {
    return __builtin_amdgcn_wmma_f32_16x16x32_bf16(
        /*neg_a=*/false, a, /*neg_b=*/false, b,
        /*c_mod=*/(short)0, c, /*reuse_a=*/false, /*reuse_b=*/false);
}

// ---------------------------------------------------------------------------
// Kernel 1: P[m,e] = (sum_d X[m,d] * W[e,d]) * scale -> bf16.
// 128x128 tile / WG, 8 waves (2x4 WMMA tiles each), K-step 32,
// double-buffered LDS, f32->bf16 conversion in registers.
// ---------------------------------------------------------------------------
__global__ __launch_bounds__(256)
void proj_kernel(const float* __restrict__ X,   // [M_, D_]
                 const float* __restrict__ W,   // [E_, D_]
                 bf16_t* __restrict__ P,        // [M_, E_] bf16
                 float scale)
{
    __shared__ __align__(32) bf16_t As[2][128][32];
    __shared__ __align__(32) bf16_t Ws[2][128][32];

    const int tid  = threadIdx.x;
    const int lane = tid & 31;
    const int wave = tid >> 5;
    const int m0   = blockIdx.x * 128;
    const int n0   = blockIdx.y * 128;
    const int wm   = (wave & 3) * 32;
    const int wn   = (wave >> 2) * 64;

    const v8f zero = {0.f, 0.f, 0.f, 0.f, 0.f, 0.f, 0.f, 0.f};
    v8f acc[2][4];
#pragma unroll
    for (int i = 0; i < 2; ++i)
#pragma unroll
        for (int j = 0; j < 4; ++j) acc[i][j] = zero;

    v4f xr[4], wr[4];

    auto load_regs = [&](int k0) {
#pragma unroll
        for (int i = 0; i < 4; ++i) {
            int slot = tid + i * 256;      // 0..1023 float4 slots
            int r = slot >> 3;
            int c = (slot & 7) * 4;
            xr[i] = *(const v4f*)(X + (size_t)(m0 + r) * D_ + k0 + c);
            wr[i] = *(const v4f*)(W + (size_t)(n0 + r) * D_ + k0 + c);
        }
    };
    auto store_lds = [&](int buf) {
#pragma unroll
        for (int i = 0; i < 4; ++i) {
            int slot = tid + i * 256;
            int r = slot >> 3;
            int c = (slot & 7) * 4;
            v4bf xb, wb;
#pragma unroll
            for (int j = 0; j < 4; ++j) { xb[j] = (bf16_t)xr[i][j]; wb[j] = (bf16_t)wr[i][j]; }
            *(v4bf*)&As[buf][r][c] = xb;
            *(v4bf*)&Ws[buf][r][c] = wb;
        }
    };
    auto compute = [&](int buf) {
        const int row = lane & 15;
        const int kh  = (lane >> 4) * 16;
        v16bf a0 = *(const v16bf*)&As[buf][wm + row][kh];
        v16bf a1 = *(const v16bf*)&As[buf][wm + 16 + row][kh];
        v16bf b0 = *(const v16bf*)&Ws[buf][wn + row][kh];
        v16bf b1 = *(const v16bf*)&Ws[buf][wn + 16 + row][kh];
        v16bf b2 = *(const v16bf*)&Ws[buf][wn + 32 + row][kh];
        v16bf b3 = *(const v16bf*)&Ws[buf][wn + 48 + row][kh];
        acc[0][0] = wmma_bf16(a0, b0, acc[0][0]);
        acc[0][1] = wmma_bf16(a0, b1, acc[0][1]);
        acc[0][2] = wmma_bf16(a0, b2, acc[0][2]);
        acc[0][3] = wmma_bf16(a0, b3, acc[0][3]);
        acc[1][0] = wmma_bf16(a1, b0, acc[1][0]);
        acc[1][1] = wmma_bf16(a1, b1, acc[1][1]);
        acc[1][2] = wmma_bf16(a1, b2, acc[1][2]);
        acc[1][3] = wmma_bf16(a1, b3, acc[1][3]);
    };

    const int NIT = D_ / 32;
    load_regs(0);
    store_lds(0);
    __syncthreads();
#pragma unroll 2
    for (int it = 0; it < NIT; ++it) {
        if (it + 1 < NIT) {
            load_regs((it + 1) * 32);
            __builtin_prefetch(X + (size_t)(m0 + (tid >> 1)) * D_ + (it + 2) * 32, 0, 1);
        }
        compute(it & 1);
        if (it + 1 < NIT) {
            store_lds((it + 1) & 1);
            __syncthreads();
        }
    }

    const int crow = (lane >> 4) * 8;
    const int ccol = lane & 15;
#pragma unroll
    for (int i = 0; i < 2; ++i)
#pragma unroll
        for (int j = 0; j < 4; ++j) {
            int mg = m0 + wm + i * 16 + crow;
            int ng = n0 + wn + j * 16 + ccol;
#pragma unroll
            for (int r2 = 0; r2 < 8; ++r2)
                P[(size_t)(mg + r2) * E_ + ng] = (bf16_t)(acc[i][j][r2] * scale);
        }
}

// ---------------------------------------------------------------------------
// Kernel 2: attn[b,q,k] = qp[b,q,:] . kp[b,k,:] (scale pre-folded into qp).
// bf16 operands are copy-through -> async global->LDS staging when available.
// ---------------------------------------------------------------------------
__global__ __launch_bounds__(256)
void scores_kernel(const bf16_t* __restrict__ qp,   // [M_, E_] bf16
                   const bf16_t* __restrict__ kp,   // [M_, E_] bf16
                   float* __restrict__ attn)        // [B_, N_, N_] f32
{
    __shared__ __align__(32) bf16_t Qs[2][128][32];
    __shared__ __align__(32) bf16_t Ks[2][128][32];

    const int tid  = threadIdx.x;
    const int lane = tid & 31;
    const int wave = tid >> 5;
    const int b    = blockIdx.z;
    const int m0   = blockIdx.x * 128;
    const int n0   = blockIdx.y * 128;
    const int wm   = (wave & 3) * 32;
    const int wn   = (wave >> 2) * 64;

    const v8f zero = {0.f, 0.f, 0.f, 0.f, 0.f, 0.f, 0.f, 0.f};
    v8f acc[2][4];
#pragma unroll
    for (int i = 0; i < 2; ++i)
#pragma unroll
        for (int j = 0; j < 4; ++j) acc[i][j] = zero;

#if HAVE_ASYNC_LDS
    auto stage = [&](int buf, int k0) {
#pragma unroll
        for (int i = 0; i < 2; ++i) {
            int slot = tid + i * 256;    // 0..511, 16B each
            int r = slot >> 2;
            int c = (slot & 3) * 8;
            __builtin_amdgcn_global_load_async_to_lds_b128(
                (gm_v4i*)(qp + ((size_t)b * N_ + m0 + r) * E_ + k0 + c),
                (lds_v4i*)&Qs[buf][r][c], 0, 0);
            __builtin_amdgcn_global_load_async_to_lds_b128(
                (gm_v4i*)(kp + ((size_t)b * N_ + n0 + r) * E_ + k0 + c),
                (lds_v4i*)&Ks[buf][r][c], 0, 0);
        }
    };
#else
    auto stage = [&](int buf, int k0) {
#pragma unroll
        for (int i = 0; i < 2; ++i) {
            int slot = tid + i * 256;
            int r = slot >> 2;
            int c = (slot & 3) * 8;
            *(v8bf*)&Qs[buf][r][c] =
                *(const v8bf*)(qp + ((size_t)b * N_ + m0 + r) * E_ + k0 + c);
            *(v8bf*)&Ks[buf][r][c] =
                *(const v8bf*)(kp + ((size_t)b * N_ + n0 + r) * E_ + k0 + c);
        }
    };
#endif

    auto compute = [&](int buf) {
        const int row = lane & 15;
        const int kh  = (lane >> 4) * 16;
        v16bf a0 = *(const v16bf*)&Qs[buf][wm + row][kh];
        v16bf a1 = *(const v16bf*)&Qs[buf][wm + 16 + row][kh];
        v16bf b0 = *(const v16bf*)&Ks[buf][wn + row][kh];
        v16bf b1 = *(const v16bf*)&Ks[buf][wn + 16 + row][kh];
        v16bf b2 = *(const v16bf*)&Ks[buf][wn + 32 + row][kh];
        v16bf b3 = *(const v16bf*)&Ks[buf][wn + 48 + row][kh];
        acc[0][0] = wmma_bf16(a0, b0, acc[0][0]);
        acc[0][1] = wmma_bf16(a0, b1, acc[0][1]);
        acc[0][2] = wmma_bf16(a0, b2, acc[0][2]);
        acc[0][3] = wmma_bf16(a0, b3, acc[0][3]);
        acc[1][0] = wmma_bf16(a1, b0, acc[1][0]);
        acc[1][1] = wmma_bf16(a1, b1, acc[1][1]);
        acc[1][2] = wmma_bf16(a1, b2, acc[1][2]);
        acc[1][3] = wmma_bf16(a1, b3, acc[1][3]);
    };

    const int NIT = E_ / 32;
    stage(0, 0);
#if HAVE_ASYNC_LDS
    wait_async0();
#endif
    __syncthreads();
#pragma unroll 2
    for (int it = 0; it < NIT; ++it) {
        if (it + 1 < NIT) stage((it + 1) & 1, (it + 1) * 32);
        compute(it & 1);
        if (it + 1 < NIT) {
#if HAVE_ASYNC_LDS
            wait_async0();
#endif
            __syncthreads();
        }
    }

    const int crow = (lane >> 4) * 8;
    const int ccol = lane & 15;
#pragma unroll
    for (int i = 0; i < 2; ++i)
#pragma unroll
        for (int j = 0; j < 4; ++j) {
            int mg = m0 + wm + i * 16 + crow;
            int ng = n0 + wn + j * 16 + ccol;
#pragma unroll
            for (int r2 = 0; r2 < 8; ++r2)
                attn[((size_t)b * N_ + mg + r2) * N_ + ng] = acc[i][j][r2];
        }
}

// ---------------------------------------------------------------------------
// Kernel 3: in-place row softmax over 2048 cols (one 256-thread block / row).
// ---------------------------------------------------------------------------
__global__ __launch_bounds__(256)
void softmax_kernel(float* __restrict__ attn)
{
    __shared__ float sm[256];
    __shared__ float ssum[256];
    const int tid = threadIdx.x;
    const size_t base = (size_t)blockIdx.x * N_;

    float m = -3.0e38f, s = 0.f;
    float vals[8];
#pragma unroll
    for (int i = 0; i < 8; ++i) {
        float x = attn[base + tid + i * 256];
        vals[i] = x;
        if (x > m) { s = s * __expf(m - x) + 1.f; m = x; }
        else       { s += __expf(x - m); }
    }
    sm[tid] = m; ssum[tid] = s;
    __syncthreads();
    for (int off = 128; off > 0; off >>= 1) {
        if (tid < off) {
            float m2 = sm[tid + off], s2 = ssum[tid + off];
            float M  = fmaxf(sm[tid], m2);
            ssum[tid] = ssum[tid] * __expf(sm[tid] - M) + s2 * __expf(m2 - M);
            sm[tid] = M;
        }
        __syncthreads();
    }
    const float M   = sm[0];
    const float inv = 1.0f / ssum[0];
#pragma unroll
    for (int i = 0; i < 8; ++i)
        attn[base + tid + i * 256] = __expf(vals[i] - M) * inv;
}

// ---------------------------------------------------------------------------
// Kernel 4: out[b,q,e] = sum_k attn[b,q,k] * vp[b,k,e].
// attn f32->bf16 at staging; vp transposed at staging; double-buffered.
// ---------------------------------------------------------------------------
__global__ __launch_bounds__(256)
void out_kernel(const float* __restrict__ attn,    // [B_, N_, N_] f32 softmaxed
                const bf16_t* __restrict__ vp,     // [M_, E_] bf16
                float* __restrict__ out)           // [B_, N_, E_] f32
{
    __shared__ __align__(32) bf16_t As[2][128][32];
    __shared__ __align__(32) bf16_t Vs[2][128][32];   // Vs[.][e][k] = vp[k][e]

    const int tid  = threadIdx.x;
    const int lane = tid & 31;
    const int wave = tid >> 5;
    const int b    = blockIdx.z;
    const int m0   = blockIdx.x * 128;
    const int n0   = blockIdx.y * 128;
    const int wm   = (wave & 3) * 32;
    const int wn   = (wave >> 2) * 64;

    const v8f zero = {0.f, 0.f, 0.f, 0.f, 0.f, 0.f, 0.f, 0.f};
    v8f acc[2][4];
#pragma unroll
    for (int i = 0; i < 2; ++i)
#pragma unroll
        for (int j = 0; j < 4; ++j) acc[i][j] = zero;

    v4f  ar[4];
    v8bf vr[2];

    auto load_regs = [&](int k0) {
#pragma unroll
        for (int i = 0; i < 4; ++i) {
            int slot = tid + i * 256;
            int r = slot >> 3;
            int c = (slot & 7) * 4;
            ar[i] = *(const v4f*)(attn + ((size_t)b * N_ + m0 + r) * N_ + k0 + c);
        }
#pragma unroll
        for (int i = 0; i < 2; ++i) {
            int slot = tid + i * 256;
            int kl = slot >> 4;
            int el = (slot & 15) * 8;
            vr[i] = *(const v8bf*)(vp + ((size_t)b * N_ + k0 + kl) * E_ + n0 + el);
        }
    };
    auto store_lds = [&](int buf) {
#pragma unroll
        for (int i = 0; i < 4; ++i) {
            int slot = tid + i * 256;
            int r = slot >> 3;
            int c = (slot & 7) * 4;
            v4bf xb;
#pragma unroll
            for (int j = 0; j < 4; ++j) xb[j] = (bf16_t)ar[i][j];
            *(v4bf*)&As[buf][r][c] = xb;
        }
#pragma unroll
        for (int i = 0; i < 2; ++i) {
            int slot = tid + i * 256;
            int kl = slot >> 4;
            int el = (slot & 15) * 8;
#pragma unroll
            for (int j = 0; j < 8; ++j) Vs[buf][el + j][kl] = vr[i][j];
        }
    };
    auto compute = [&](int buf) {
        const int row = lane & 15;
        const int kh  = (lane >> 4) * 16;
        v16bf a0 = *(const v16bf*)&As[buf][wm + row][kh];
        v16bf a1 = *(const v16bf*)&As[buf][wm + 16 + row][kh];
        v16bf b0 = *(const v16bf*)&Vs[buf][wn + row][kh];
        v16bf b1 = *(const v16bf*)&Vs[buf][wn + 16 + row][kh];
        v16bf b2 = *(const v16bf*)&Vs[buf][wn + 32 + row][kh];
        v16bf b3 = *(const v16bf*)&Vs[buf][wn + 48 + row][kh];
        acc[0][0] = wmma_bf16(a0, b0, acc[0][0]);
        acc[0][1] = wmma_bf16(a0, b1, acc[0][1]);
        acc[0][2] = wmma_bf16(a0, b2, acc[0][2]);
        acc[0][3] = wmma_bf16(a0, b3, acc[0][3]);
        acc[1][0] = wmma_bf16(a1, b0, acc[1][0]);
        acc[1][1] = wmma_bf16(a1, b1, acc[1][1]);
        acc[1][2] = wmma_bf16(a1, b2, acc[1][2]);
        acc[1][3] = wmma_bf16(a1, b3, acc[1][3]);
    };

    const int NIT = N_ / 32;   // 64
    load_regs(0);
    store_lds(0);
    __syncthreads();
    for (int it = 0; it < NIT; ++it) {
        if (it + 1 < NIT) load_regs((it + 1) * 32);
        compute(it & 1);
        if (it + 1 < NIT) {
            store_lds((it + 1) & 1);
            __syncthreads();
        }
    }

    const int crow = (lane >> 4) * 8;
    const int ccol = lane & 15;
#pragma unroll
    for (int i = 0; i < 2; ++i)
#pragma unroll
        for (int j = 0; j < 4; ++j) {
            int mg = m0 + wm + i * 16 + crow;
            int ng = n0 + wn + j * 16 + ccol;
#pragma unroll
            for (int r2 = 0; r2 < 8; ++r2)
                out[((size_t)b * N_ + mg + r2) * E_ + ng] = acc[i][j][r2];
        }
}

// ---------------------------------------------------------------------------
// Host-side launch sequence (all on `stream`, deterministic, capture-safe).
// ---------------------------------------------------------------------------
extern "C" void kernel_launch(void* const* d_in, const int* in_sizes, int n_in,
                              void* d_out, int out_size, void* d_ws, size_t ws_size,
                              hipStream_t stream)
{
    (void)in_sizes; (void)n_in; (void)out_size; (void)ws_size;

    const float* q  = (const float*)d_in[0];
    const float* k  = (const float*)d_in[1];
    const float* v  = (const float*)d_in[2];
    const float* Wq = (const float*)d_in[3];
    const float* Wk = (const float*)d_in[4];
    const float* Wv = (const float*)d_in[5];

    float* out  = (float*)d_out;                        // [B,N,E]
    float* attn = out + (size_t)B_ * N_ * E_;           // [B,N,N]

    char* ws = (char*)d_ws;
    const size_t projBytes = (size_t)M_ * E_ * sizeof(bf16_t);   // 32 MB each
    bf16_t* qp = (bf16_t*)(ws);
    bf16_t* kp = (bf16_t*)(ws + projBytes);
    bf16_t* vp = (bf16_t*)(ws + 2 * projBytes);

    const float cD = 1.0f / sqrtf((float)D_);   // equalized-LR factor == attn scale

    dim3 blk(256);
    dim3 gp(M_ / 128, E_ / 128, 1);             // (256, 4)
    proj_kernel<<<gp, blk, 0, stream>>>(q, Wq, qp, cD * cD);  // + attn scale folded
    proj_kernel<<<gp, blk, 0, stream>>>(k, Wk, kp, cD);
    proj_kernel<<<gp, blk, 0, stream>>>(v, Wv, vp, cD);

    dim3 gs(N_ / 128, N_ / 128, B_);            // (16, 16, 16)
    scores_kernel<<<gs, blk, 0, stream>>>(qp, kp, attn);

    softmax_kernel<<<dim3(B_ * N_), blk, 0, stream>>>(attn);

    dim3 go(N_ / 128, E_ / 128, B_);            // (16, 4, 16)
    out_kernel<<<go, blk, 0, stream>>>(attn, vp, out);
}
